// SingleHeadAttention_83734682402984
// MI455X (gfx1250) — compile-verified, matrix operand order checked by reading, not run on hardware
//
#include <hip/hip_runtime.h>

// Problem constants (match reference)
#define B_  4
#define T_  4096
#define D_  1024
#define A_  1024

typedef unsigned short u16;
typedef __attribute__((ext_vector_type(16))) __bf16 v16bf;
typedef __attribute__((ext_vector_type(8)))  float  v8f;

// ---------------------------------------------------------------------------
// Async global->LDS copy (CDNA5 ASYNCcnt path), with sync fallback.
// clang prototype: void(v4i addrspace(1)* src, v4i addrspace(3)* dst,
//                       Imm int offset, Imm int cpol)
// ---------------------------------------------------------------------------
#if defined(__has_builtin)
# if __has_builtin(__builtin_amdgcn_global_load_async_to_lds_b128)
#  define HAVE_ASYNC 1
# endif
#endif
#ifndef HAVE_ASYNC
# define HAVE_ASYNC 0
#endif

typedef int v4i __attribute__((vector_size(16)));
typedef __attribute__((address_space(1))) v4i* gptr_b128;
typedef __attribute__((address_space(3))) v4i* lptr_b128;

__device__ __forceinline__ void cp16(const u16* g, u16* l) {
#if HAVE_ASYNC
  __builtin_amdgcn_global_load_async_to_lds_b128((gptr_b128)g, (lptr_b128)l, 0, 0);
#else
  *reinterpret_cast<uint4*>(l) = *reinterpret_cast<const uint4*>(g);
#endif
}

__device__ __forceinline__ void wait_async() {
#if HAVE_ASYNC
# if defined(__has_builtin) && __has_builtin(__builtin_amdgcn_s_wait_asynccnt)
  __builtin_amdgcn_s_wait_asynccnt(0);
# else
  asm volatile("s_wait_asynccnt 0x0" ::: "memory");
# endif
#endif
}

// ---------------------------------------------------------------------------
union FragU { uint4 q[2]; v16bf v; };

// Load one 16x32 bf16 WMMA operand fragment (ISA 7.12.2 16-bit layout):
// per-lane 8 contiguous bf16 at +0 and 8 more at +16 elements.
__device__ __forceinline__ v16bf load_frag(const u16* p) {
  FragU f;
  f.q[0] = *reinterpret_cast<const uint4*>(p);
  f.q[1] = *reinterpret_cast<const uint4*>(p + 16);
  return f.v;
}

__device__ __forceinline__ v8f wmma_bf16(v16bf a, v16bf b, v8f c) {
  return __builtin_amdgcn_wmma_f32_16x16x32_bf16(false, a, false, b, (short)0, c, false, false);
}

__device__ __forceinline__ u16 f2bf(float x) {  // round-to-nearest-even
  unsigned u = __float_as_uint(x);
  u += 0x7FFFu + ((u >> 16) & 1u);
  return (u16)(u >> 16);
}

__device__ __forceinline__ v8f zero8() {
  v8f z;
#pragma unroll
  for (int i = 0; i < 8; ++i) z[i] = 0.0f;
  return z;
}

// ---------------------------------------------------------------------------
// Kernel 0: fp32 -> bf16 conversion (grid-stride)
// ---------------------------------------------------------------------------
__global__ void cvt_f32_bf16(const float* __restrict__ src, u16* __restrict__ dst, int n) {
  int i = blockIdx.x * blockDim.x + threadIdx.x;
  int stride = gridDim.x * blockDim.x;
  for (; i < n; i += stride) dst[i] = f2bf(src[i]);
}

// ---------------------------------------------------------------------------
// Kernel 1: QKV projection GEMM, LDS double-buffered with async copies.
// 8-wave workgroup computes a 128(M) x 256(N) tile; waves arranged 2x4, each
// wave owns a 64x64 block (4x4 WMMA tiles).  Per 32-wide k-step we stage
// X(128x32) + W(256x32) bf16 tiles (24KB) into the other LDS buffer with
// global_load_async_to_lds_b128 while WMMAs consume the current buffer.
// blockIdx.z selects Q / K / V; V is stored TRANSPOSED ([A,T] per batch).
// ---------------------------------------------------------------------------
__global__ __launch_bounds__(256) void qkv_gemm(
    const u16* __restrict__ X,
    const u16* __restrict__ Wq, const u16* __restrict__ Wk, const u16* __restrict__ Wv,
    u16* __restrict__ Qo, u16* __restrict__ Ko, u16* __restrict__ Vt) {
  __shared__ __align__(16) u16 Xs[2][128 * 32];   // 2 x 8KB
  __shared__ __align__(16) u16 Ws[2][256 * 32];   // 2 x 16KB

  const int tid  = threadIdx.x;
  const int lane = tid & 31;
  const int lr   = lane & 15;
  const int hi   = (lane >> 4) & 1;
  const int w    = tid >> 5;           // wave 0..7
  const int wm   = (w >> 2) & 1;       // 0..1  (64-row slice)
  const int wn   = w & 3;              // 0..3  (64-col slice)

  const int m0 = blockIdx.x * 128;
  const int n0 = blockIdx.y * 256;
  const int which = blockIdx.z;
  const u16* W = (which == 0) ? Wq : (which == 1) ? Wk : Wv;

  v8f c[4][4];
#pragma unroll
  for (int mi = 0; mi < 4; ++mi)
#pragma unroll
    for (int ni = 0; ni < 4; ++ni) c[mi][ni] = zero8();

  // ---- stage tiles for one k-step into buffer `buf` ----
  auto stage = [&](int buf, int kk) {
    // X tile: 128 rows x 32 cols = 512 16B chunks, 2 per thread
#pragma unroll
    for (int cth = 0; cth < 2; ++cth) {
      int chunk = tid * 2 + cth;
      int row = chunk >> 2, part = chunk & 3;
      cp16(X + (size_t)(m0 + row) * D_ + kk + part * 8,
           &Xs[buf][row * 32 + part * 8]);
    }
    // W tile: 256 rows x 32 cols = 1024 16B chunks, 4 per thread
#pragma unroll
    for (int cth = 0; cth < 4; ++cth) {
      int chunk = tid * 4 + cth;
      int row = chunk >> 2, part = chunk & 3;
      cp16(W + (size_t)(n0 + row) * D_ + kk + part * 8,
           &Ws[buf][row * 32 + part * 8]);
    }
  };

  stage(0, 0);
  wait_async();
  __syncthreads();

  int buf = 0;
  for (int kk = 0; kk < D_; kk += 32) {
    if (kk + 32 < D_) stage(buf ^ 1, kk + 32);   // overlap copy with compute

    v16bf a[4], bm[4];
#pragma unroll
    for (int mi = 0; mi < 4; ++mi)
      a[mi] = load_frag(&Xs[buf][(wm * 64 + mi * 16 + lr) * 32 + hi * 8]);
#pragma unroll
    for (int ni = 0; ni < 4; ++ni)
      bm[ni] = load_frag(&Ws[buf][(wn * 64 + ni * 16 + lr) * 32 + hi * 8]);
#pragma unroll
    for (int mi = 0; mi < 4; ++mi)
#pragma unroll
      for (int ni = 0; ni < 4; ++ni)
        c[mi][ni] = wmma_bf16(a[mi], bm[ni], c[mi][ni]);

    wait_async();
    __syncthreads();
    buf ^= 1;
  }

  const int m0w = m0 + wm * 64;
  const int n0w = n0 + wn * 64;
  if (which < 2) {
    u16* out = (which == 0) ? Qo : Ko;
#pragma unroll
    for (int mi = 0; mi < 4; ++mi)
#pragma unroll
      for (int ni = 0; ni < 4; ++ni)
#pragma unroll
        for (int i = 0; i < 8; ++i)
          out[(size_t)(m0w + mi * 16 + hi * 8 + i) * A_ + (n0w + ni * 16 + lr)] =
              f2bf(c[mi][ni][i]);
  } else {
    // V transposed: Vt[b][a][t]; 8 consecutive t per lane -> one b128 store
    const int b  = m0w >> 12;
    const int t0 = m0w & (T_ - 1);
#pragma unroll
    for (int mi = 0; mi < 4; ++mi)
#pragma unroll
      for (int ni = 0; ni < 4; ++ni) {
        const int a = n0w + ni * 16 + lr;
        const int t = t0 + mi * 16 + hi * 8;
        u16 pk[8];
#pragma unroll
        for (int i = 0; i < 8; ++i) pk[i] = f2bf(c[mi][ni][i]);
        *reinterpret_cast<uint4*>(Vt + ((size_t)b * A_ + a) * T_ + t) =
            *reinterpret_cast<const uint4*>(pk);
      }
  }
}

// ---------------------------------------------------------------------------
// Kernel 2: streaming (flash) causal attention.
// 16-wave workgroup, 64 query rows.  Per 256-key block:
//   phase1: wave w computes S slice (64 x 16 keys) via 128 WMMAs -> LDS
//   phase2: online softmax stats (mask, rowmax, exp, rowsum) in LDS
//   phase3: rescale O accumulators, O += P x V (128 WMMAs per wave; each
//           wave owns a 64-wide head-dim slice of V^T)
// ---------------------------------------------------------------------------
__global__ __launch_bounds__(512) void attn_kernel(
    const u16* __restrict__ Q, const u16* __restrict__ K,
    const u16* __restrict__ Vt, float* __restrict__ Out) {
  __shared__ __align__(16) float Sm[64 * 256];   // raw scores (f32)
  __shared__ __align__(16) u16   Pm[64 * 256];   // exp'd probs (bf16)
  __shared__ float rowM[64], rowL[64], rowFac[64];

  const int tid  = threadIdx.x;
  const int lane = tid & 31;
  const int lr   = lane & 15;
  const int hi   = (lane >> 4) & 1;
  const int w    = tid >> 5;            // wave 0..15
  const int q0   = blockIdx.x * 64;
  const int b    = blockIdx.y;
  const int a0   = w * 64;              // this wave's head-dim slice

  const u16* Qb = Q  + (size_t)b * T_ * A_;
  const u16* Kb = K  + (size_t)b * T_ * A_;
  const u16* Vb = Vt + (size_t)b * A_ * T_;
  float*     Ob = Out + (size_t)b * T_ * A_;

  v8f o[4][4];
#pragma unroll
  for (int mi = 0; mi < 4; ++mi)
#pragma unroll
    for (int ni = 0; ni < 4; ++ni) o[mi][ni] = zero8();

  if (tid < 64) { rowM[tid] = -1e30f; rowL[tid] = 0.0f; }
  __syncthreads();

  const int nkb = (q0 >> 8) + 1;        // causal: key blocks 0 .. q0/256
  for (int kb = 0; kb < nkb; ++kb) {
    const int k0 = kb * 256;

    // prefetch next key block's K/V tiles into cache (global_prefetch_b8)
    if (kb + 1 < nkb) {
      __builtin_prefetch(Kb + (size_t)(k0 + 256 + w * 16 + lr) * A_, 0, 3);
      __builtin_prefetch(Vb + (size_t)(a0 + lr) * T_ + k0 + 256, 0, 3);
    }

    // ---- phase 1: S = Q K^T / sqrt(A) for this wave's 16-key slice ----
    {
      const int key0 = k0 + w * 16;
      v8f s[4];
#pragma unroll
      for (int mi = 0; mi < 4; ++mi) s[mi] = zero8();
      for (int kk = 0; kk < D_; kk += 32) {
        v16bf kf = load_frag(Kb + (size_t)(key0 + lr) * A_ + kk + hi * 8);
#pragma unroll
        for (int mi = 0; mi < 4; ++mi) {
          v16bf qf = load_frag(Qb + (size_t)(q0 + mi * 16 + lr) * A_ + kk + hi * 8);
          s[mi] = wmma_bf16(qf, kf, s[mi]);
        }
      }
#pragma unroll
      for (int mi = 0; mi < 4; ++mi)
#pragma unroll
        for (int i = 0; i < 8; ++i)
          Sm[(mi * 16 + hi * 8 + i) * 256 + w * 16 + lr] = s[mi][i] * 0.03125f;
    }
    __syncthreads();

    // ---- phase 2: causal mask + online softmax stats ----
    {
      const int r   = tid >> 3;        // row 0..63
      const int sub = tid & 7;         // 8 threads per row, 32 cols each
      const int q   = q0 + r;
      const int c0  = sub * 32;
      const float mOld = rowM[r];
      const float lOld = rowL[r];

      float bmax = -1e30f;
#pragma unroll 4
      for (int j = 0; j < 32; ++j) {
        float s = Sm[r * 256 + c0 + j];
        s = (k0 + c0 + j <= q) ? s : -1e30f;
        bmax = fmaxf(bmax, s);
      }
#pragma unroll
      for (int off = 1; off < 8; off <<= 1)
        bmax = fmaxf(bmax, __shfl_xor(bmax, off, 8));
      const float mNew = fmaxf(mOld, bmax);

      float sum = 0.0f;
#pragma unroll 4
      for (int j = 0; j < 32; ++j) {
        float s = Sm[r * 256 + c0 + j];
        s = (k0 + c0 + j <= q) ? s : -1e30f;
        float e = __expf(s - mNew);
        Pm[r * 256 + c0 + j] = f2bf(e);
        sum += e;
      }
#pragma unroll
      for (int off = 1; off < 8; off <<= 1)
        sum += __shfl_xor(sum, off, 8);

      if (sub == 0) {
        const float fac = __expf(mOld - mNew);
        rowFac[r] = fac;
        rowM[r]   = mNew;
        rowL[r]   = lOld * fac + sum;
      }
    }
    __syncthreads();

    // ---- phase 3: rescale O, then O += P x V (this wave's A-slice) ----
    {
#pragma unroll
      for (int mi = 0; mi < 4; ++mi)
#pragma unroll
        for (int i = 0; i < 8; ++i) {
          const float f = rowFac[mi * 16 + hi * 8 + i];
#pragma unroll
          for (int ni = 0; ni < 4; ++ni) o[mi][ni][i] *= f;
        }
#pragma unroll
      for (int kt = 0; kt < 8; ++kt) {
        v16bf vf[4];
#pragma unroll
        for (int ni = 0; ni < 4; ++ni)
          vf[ni] = load_frag(Vb + (size_t)(a0 + ni * 16 + lr) * T_ + k0 + kt * 32 + hi * 8);
#pragma unroll
        for (int mi = 0; mi < 4; ++mi) {
          v16bf pf = load_frag(&Pm[(mi * 16 + lr) * 256 + kt * 32 + hi * 8]);
#pragma unroll
          for (int ni = 0; ni < 4; ++ni)
            o[mi][ni] = wmma_bf16(pf, vf[ni], o[mi][ni]);
        }
      }
    }
    __syncthreads();
  }

  // ---- epilogue: O / rowL -> fp32 output ----
#pragma unroll
  for (int mi = 0; mi < 4; ++mi)
#pragma unroll
    for (int i = 0; i < 8; ++i) {
      const int r = mi * 16 + hi * 8 + i;
      const float inv = 1.0f / rowL[r];
#pragma unroll
      for (int ni = 0; ni < 4; ++ni)
        Ob[(size_t)(q0 + r) * A_ + a0 + ni * 16 + lr] = o[mi][ni][i] * inv;
    }
}

// ---------------------------------------------------------------------------
extern "C" void kernel_launch(void* const* d_in, const int* in_sizes, int n_in,
                              void* d_out, int out_size, void* d_ws, size_t ws_size,
                              hipStream_t stream) {
  const float* emb = (const float*)d_in[0];   // [B,T,D]
  const float* Wq  = (const float*)d_in[1];   // [A,D]
  const float* Wk  = (const float*)d_in[2];
  const float* Wv  = (const float*)d_in[3];
  float* out = (float*)d_out;                 // [B,T,A]

  const size_t nEmb = (size_t)B_ * T_ * D_;   // 16,777,216
  const size_t nW   = (size_t)A_ * D_;        // 1,048,576
  const size_t nQKV = (size_t)B_ * T_ * A_;   // 16,777,216

  u16* ws   = (u16*)d_ws;
  u16* embB = ws;
  u16* WqB  = embB + nEmb;
  u16* WkB  = WqB + nW;
  u16* WvB  = WkB + nW;
  u16* Qb   = WvB + nW;
  u16* Kb   = Qb + nQKV;
  u16* Vt   = Kb + nQKV;                      // [B, A, T]

  // Stage 0: fp32 -> bf16 (then everything below lives in the 192MB L2)
  cvt_f32_bf16<<<4096, 256, 0, stream>>>(emb, embB, (int)nEmb);
  cvt_f32_bf16<<<1024, 256, 0, stream>>>(Wq, WqB, (int)nW);
  cvt_f32_bf16<<<1024, 256, 0, stream>>>(Wk, WkB, (int)nW);
  cvt_f32_bf16<<<1024, 256, 0, stream>>>(Wv, WvB, (int)nW);

  // Stage 1: QKV projections (z: 0=Q, 1=K, 2=V-transposed)
  qkv_gemm<<<dim3((B_ * T_) / 128, A_ / 256, 3), 256, 0, stream>>>(
      embB, WqB, WkB, WvB, Qb, Kb, Vt);

  // Stage 2: fused causal flash attention
  attn_kernel<<<dim3(T_ / 64, B_), 512, 0, stream>>>(Qb, Kb, Vt, out);
}
